// Attention_42777874268408
// MI455X (gfx1250) — matrix operand order repaired and checked by reading, not compiled, not run on hardware
//
#include <hip/hip_runtime.h>
#include <cstdint>

// ---------------------------------------------------------------------------
// Problem constants (match reference)
// ---------------------------------------------------------------------------
#define BATCH  4
#define NPTS   1024
#define DIMF   256
#define HEADS  8
#define DHEAD  64
#define INNERD 512
#define QKVN   1536
#define DYDIM  16

typedef __bf16 bf16;
typedef bf16  v16bf __attribute__((ext_vector_type(16)));
typedef bf16  v8bf  __attribute__((ext_vector_type(8)));
typedef float v8f   __attribute__((ext_vector_type(8)));

// D = A(16x32 bf16) * B(32x16 bf16) + C(16x16 f32)
__device__ __forceinline__ v8f wmma_bf16(v16bf a, v16bf b, v8f c) {
  return __builtin_amdgcn_wmma_f32_16x16x32_bf16(false, a, false, b, (short)0, c,
                                                 false, false);
}

// A-fragment (16x32, MxK) from row-major [*, ld] bf16.
// ISA layout: lanes 0-15 hold M=lane, K={0..7,16..23}; lanes 16-31 M=lane-16,
// K={8..15,24..31} -> per lane two contiguous 16B chunks.
__device__ __forceinline__ v16bf load_A16x32(const bf16* __restrict__ p, int ld,
                                             int m0, int k0, int lane) {
  const int m  = m0 + (lane & 15);
  const int lo = k0 + ((lane & 16) ? 8 : 0);
  union { v16bf v; v8bf h[2]; } u;
  u.h[0] = *(const v8bf*)(p + (size_t)m * ld + lo);
  u.h[1] = *(const v8bf*)(p + (size_t)m * ld + lo + 16);
  return u.v;
}

// B-fragment (32x16, KxN) read from "B-transposed" storage bT[n][k] (ld = K
// stride). Lane holds column n = n0 + lane%16; K={0..7,16..23} for lanes 0-15,
// K={8..15,24..31} for lanes 16-31 -> two contiguous 16B chunks per lane.
__device__ __forceinline__ v16bf load_B32x16(const bf16* __restrict__ p, int ld,
                                             int n0, int k0, int lane) {
  const int n = n0 + (lane & 15);
  const int k = k0 + ((lane & 16) ? 8 : 0);
  union { v16bf v; v8bf h[2]; } u;
  u.h[0] = *(const v8bf*)(p + (size_t)n * ld + k);
  u.h[1] = *(const v8bf*)(p + (size_t)n * ld + k + 16);
  return u.v;
}

__device__ __forceinline__ float gelu_erf(float x) {
  return 0.5f * x * (1.0f + erff(x * 0.70710678118654752f));
}
__device__ __forceinline__ float leakyf(float x) {
  return x >= 0.0f ? x : 0.01f * x;
}

// ---------------------------------------------------------------------------
// Weight converts: build bf16, N-major (transposed) copies for WMMA B loads
// ---------------------------------------------------------------------------
__global__ void cvt_wqkv_kernel(const float* __restrict__ w, bf16* __restrict__ wT) {
  const int idx = blockIdx.x * 256 + threadIdx.x;   // 1536*256
  const int n = idx >> 8, k = idx & 255;            // wT[n][k], w is (256,1536)
  wT[idx] = (bf16)w[(size_t)k * QKVN + n];
}

__global__ void cvt_wout_kernel(const float* __restrict__ w, bf16* __restrict__ wT) {
  const int idx = blockIdx.x * 256 + threadIdx.x;   // 256*512
  const int n = idx >> 9, k = idx & 511;            // wT[n][k], w is (512,256)
  wT[idx] = (bf16)w[(size_t)k * DIMF + n];
}

// ---------------------------------------------------------------------------
// LayerNorm -> bf16 x_hat (4096 x 256), one block per row
// ---------------------------------------------------------------------------
__global__ void ln_kernel(const float* __restrict__ x, const float* __restrict__ gamma,
                          const float* __restrict__ beta, bf16* __restrict__ xn) {
  __shared__ float s1[256];
  __shared__ float s2[256];
  const int row = blockIdx.x, t = threadIdx.x;
  const float v = x[(size_t)row * DIMF + t];
  s1[t] = v; s2[t] = v * v;
  __syncthreads();
  for (int off = 128; off > 0; off >>= 1) {
    if (t < off) { s1[t] += s1[t + off]; s2[t] += s2[t + off]; }
    __syncthreads();
  }
  const float mu  = s1[0] * (1.0f / DIMF);
  const float var = s2[0] * (1.0f / DIMF) - mu * mu;
  const float inv = rsqrtf(var + 1e-5f);
  xn[(size_t)row * DIMF + t] = (bf16)((v - mu) * inv * gamma[t] + beta[t]);
}

// ---------------------------------------------------------------------------
// Pairwise modulated delta: computed ONCE (shared by all 8 heads).
// out[b,i,j,c] = leaky((pos*neg)@Wc + bc)[c] * (xyz[b,j]-xyz[b,i])[c]
// ---------------------------------------------------------------------------
__global__ void delta_kernel(const float* __restrict__ xyzs,
                             const float* __restrict__ Wp1, const float* __restrict__ Wp2,
                             const float* __restrict__ Wn1, const float* __restrict__ Wn2,
                             const float* __restrict__ Wc,  const float* __restrict__ bc,
                             float* __restrict__ dmod) {
  const size_t tid = (size_t)blockIdx.x * 256 + threadIdx.x;   // 4M pairs
  const int j = (int)(tid & (NPTS - 1));
  const int i = (int)((tid >> 10) & (NPTS - 1));
  const int b = (int)(tid >> 20);
  const float* pj = xyzs + ((size_t)b * NPTS + j) * 3;
  const float* pi = xyzs + ((size_t)b * NPTS + i) * 3;
  const float d0 = pj[0] - pi[0], d1 = pj[1] - pi[1], d2 = pj[2] - pi[2];

  const float tp0 = fmaxf(d0, 0.f), tp1 = fmaxf(d1, 0.f), tp2 = fmaxf(d2, 0.f);
  const float tn0 = fmaxf(-d0, 0.f), tn1 = fmaxf(-d1, 0.f), tn2 = fmaxf(-d2, 0.f);
  float p0 = 0.f, p1 = 0.f, p2 = 0.f, q0 = 0.f, q1 = 0.f, q2 = 0.f;
#pragma unroll
  for (int u = 0; u < DYDIM; ++u) {
    const float hp = tp0 * Wp1[u] + tp1 * Wp1[16 + u] + tp2 * Wp1[32 + u];
    const float gp = gelu_erf(hp);
    p0 += gp * Wp2[u * 3 + 0]; p1 += gp * Wp2[u * 3 + 1]; p2 += gp * Wp2[u * 3 + 2];
    const float hn = tn0 * Wn1[u] + tn1 * Wn1[16 + u] + tn2 * Wn1[32 + u];
    const float gn = gelu_erf(hn);
    q0 += gn * Wn2[u * 3 + 0]; q1 += gn * Wn2[u * 3 + 1]; q2 += gn * Wn2[u * 3 + 2];
  }
  p0 = leakyf(p0); p1 = leakyf(p1); p2 = leakyf(p2);
  q0 = leakyf(q0); q1 = leakyf(q1); q2 = leakyf(q2);
  const float e0 = p0 * q0, e1 = p1 * q1, e2 = p2 * q2;
  const float c0 = leakyf(e0 * Wc[0] + e1 * Wc[3] + e2 * Wc[6] + bc[0]);
  const float c1 = leakyf(e0 * Wc[1] + e1 * Wc[4] + e2 * Wc[7] + bc[1]);
  const float c2 = leakyf(e0 * Wc[2] + e1 * Wc[5] + e2 * Wc[8] + bc[2]);
  float* o = dmod + tid * 3;
  o[0] = c0 * d0; o[1] = c1 * d1; o[2] = c2 * d2;
}

// ---------------------------------------------------------------------------
// QKV GEMM (4096x256 @ 256x1536), WMMA bf16. One 16x16 tile per wave.
// Q,K stored [b*h, n, d]; V stored TRANSPOSED [b*h, d, n] for contiguous
// B-fragments in the P@V stage.
// ---------------------------------------------------------------------------
__global__ void qkv_gemm_kernel(const bf16* __restrict__ xn, const bf16* __restrict__ wT,
                                bf16* __restrict__ q, bf16* __restrict__ k,
                                bf16* __restrict__ vt) {
  const int lane = threadIdx.x & 31;
  const int wave = threadIdx.x >> 5;
  const int tile = blockIdx.x * 8 + wave;     // 256*96 tiles
  const int NT = QKVN / 16;                   // 96
  const int m0 = (tile / NT) * 16;
  const int n0 = (tile % NT) * 16;
  v8f acc = {};
#pragma unroll
  for (int kk = 0; kk < DIMF; kk += 32)
    acc = wmma_bf16(load_A16x32(xn, DIMF, m0, kk, lane),
                    load_B32x16(wT, DIMF, n0, kk, lane), acc);
  const int sect = n0 / INNERD;                  // 0=q, 1=k, 2=v
  const int h    = (n0 % INNERD) / DHEAD;
  const int d    = (n0 % DHEAD) + (lane & 15);
  const int mof  = (lane & 16) ? 8 : 0;
#pragma unroll
  for (int r = 0; r < 8; ++r) {
    const int m = m0 + r + mof;
    const int b = m >> 10, i = m & (NPTS - 1);
    const bf16 v = (bf16)acc[r];
    const size_t bh = (size_t)(b * HEADS + h);
    if (sect == 0)      q [(bh * NPTS + i) * DHEAD + d] = v;
    else if (sect == 1) k [(bh * NPTS + i) * DHEAD + d] = v;
    else                vt[(bh * DHEAD + d) * NPTS + i] = v;
  }
}

// ---------------------------------------------------------------------------
// Fused flash-attention + attention-weighted delta reduction + spatial proj.
// One wave handles (b, h, 16 queries); online softmax over 1024 keys in
// 32-key chunks. P goes through LDS to convert C-layout f32 -> A-layout bf16.
// ---------------------------------------------------------------------------
__global__ void attn_kernel(const bf16* __restrict__ q, const bf16* __restrict__ k,
                            const bf16* __restrict__ vt, const float* __restrict__ dmod,
                            const float* __restrict__ Wsp, bf16* __restrict__ ao) {
  __shared__ __align__(32) bf16 plds[8][16 * 32];
  const int lane = threadIdx.x & 31;
  const int wave = threadIdx.x >> 5;
  const int task = blockIdx.x * 8 + wave;        // 4*8*64 = 2048 tasks
  const int it = task & 63;
  const int bh = task >> 6;
  const int b  = bh >> 3;
  const int h  = bh & 7;
  const int i0 = it * 16;
  const bf16* qh = q  + (size_t)bh * NPTS * DHEAD;
  const bf16* kh = k  + (size_t)bh * NPTS * DHEAD;
  const bf16* vh = vt + (size_t)bh * DHEAD * NPTS;
  const int mof = (lane & 16) ? 8 : 0;
  const int nn  = lane & 15;

  const v16bf aq0 = load_A16x32(qh, DHEAD, i0, 0,  lane);
  const v16bf aq1 = load_A16x32(qh, DHEAD, i0, 32, lane);

  v8f oacc[4] = {{}, {}, {}, {}};
  float mrow[8], lrow[8], wd0[8], wd1[8], wd2[8];
#pragma unroll
  for (int r = 0; r < 8; ++r) {
    mrow[r] = -3.0e38f; lrow[r] = 0.f; wd0[r] = wd1[r] = wd2[r] = 0.f;
  }

  for (int j0 = 0; j0 < NPTS; j0 += 32) {
    v8f c0 = {}, c1 = {};
    c0 = wmma_bf16(aq0, load_B32x16(kh, DHEAD, j0,      0,  lane), c0);
    c0 = wmma_bf16(aq1, load_B32x16(kh, DHEAD, j0,      32, lane), c0);
    c1 = wmma_bf16(aq0, load_B32x16(kh, DHEAD, j0 + 16, 0,  lane), c1);
    c1 = wmma_bf16(aq1, load_B32x16(kh, DHEAD, j0 + 16, 32, lane), c1);

#pragma unroll
    for (int r = 0; r < 8; ++r) {
      const float s0 = c0[r] * 0.125f;        // scale = dhead^-0.5
      const float s1 = c1[r] * 0.125f;
      float mx = fmaxf(s0, s1);
#pragma unroll
      for (int off = 1; off < 16; off <<= 1) mx = fmaxf(mx, __shfl_xor(mx, off, 16));
      const float mnew = fmaxf(mrow[r], mx);
      const float corr = __expf(mrow[r] - mnew);
      mrow[r] = mnew;
      const float p0 = __expf(s0 - mnew);
      const float p1 = __expf(s1 - mnew);
      float rs = p0 + p1;
#pragma unroll
      for (int off = 1; off < 16; off <<= 1) rs += __shfl_xor(rs, off, 16);
      lrow[r] = lrow[r] * corr + rs;
      oacc[0][r] *= corr; oacc[1][r] *= corr; oacc[2][r] *= corr; oacc[3][r] *= corr;
      // stage P for the C->A transpose
      plds[wave][(r + mof) * 32 + nn]      = (bf16)p0;
      plds[wave][(r + mof) * 32 + 16 + nn] = (bf16)p1;
      // fused attention-weighted delta reduction (i appears on both sides of
      // the einsum, so this is per-row FMA work, not a matmul)
      const size_t irow = (size_t)b * NPTS + i0 + r + mof;
      const float* dp = dmod + (irow * NPTS + j0 + nn) * 3;
      wd0[r] = wd0[r] * corr + p0 * dp[0] + p1 * dp[48];
      wd1[r] = wd1[r] * corr + p0 * dp[1] + p1 * dp[49];
      wd2[r] = wd2[r] * corr + p0 * dp[2] + p1 * dp[50];
    }

    // Read P back in A-layout (same-wave DS ops are in-order; the compiler
    // inserts the s_wait_dscnt for the load results).
    union { v16bf v; v8bf hh[2]; } ua;
    {
      const int m  = nn;
      const int lo = (lane & 16) ? 8 : 0;
      ua.hh[0] = *(const v8bf*)&plds[wave][m * 32 + lo];
      ua.hh[1] = *(const v8bf*)&plds[wave][m * 32 + lo + 16];
    }
    oacc[0] = wmma_bf16(ua.v, load_B32x16(vh, NPTS, 0,  j0, lane), oacc[0]);
    oacc[1] = wmma_bf16(ua.v, load_B32x16(vh, NPTS, 16, j0, lane), oacc[1]);
    oacc[2] = wmma_bf16(ua.v, load_B32x16(vh, NPTS, 32, j0, lane), oacc[2]);
    oacc[3] = wmma_bf16(ua.v, load_B32x16(vh, NPTS, 48, j0, lane), oacc[3]);
  }

  // normalize; butterfly-reduce wdelta across the 16 key-columns per row
#pragma unroll
  for (int r = 0; r < 8; ++r) {
    const float inv = 1.0f / lrow[r];
    oacc[0][r] *= inv; oacc[1][r] *= inv; oacc[2][r] *= inv; oacc[3][r] *= inv;
    float a0 = wd0[r] * inv, a1 = wd1[r] * inv, a2 = wd2[r] * inv;
#pragma unroll
    for (int off = 1; off < 16; off <<= 1) {
      a0 += __shfl_xor(a0, off, 16);
      a1 += __shfl_xor(a1, off, 16);
      a2 += __shfl_xor(a2, off, 16);
    }
    wd0[r] = a0; wd1[r] = a1; wd2[r] = a2;
  }

  // fold spatial projection (wdelta @ W_spatial) into the store epilogue
#pragma unroll
  for (int t = 0; t < 4; ++t) {
    const int d = t * 16 + nn;
    const float w0 = Wsp[d], w1 = Wsp[64 + d], w2 = Wsp[128 + d];
#pragma unroll
    for (int r = 0; r < 8; ++r) {
      const int irow = i0 + r + mof;
      const float val = oacc[t][r] + wd0[r] * w0 + wd1[r] * w1 + wd2[r] * w2;
      ao[((size_t)(b * NPTS + irow)) * INNERD + h * DHEAD + d] = (bf16)val;
    }
  }
}

// ---------------------------------------------------------------------------
// Out projection (4096x512 @ 512x256) + bias + erf-GELU + residual -> f32
// ---------------------------------------------------------------------------
__global__ void out_gemm_kernel(const bf16* __restrict__ ao, const bf16* __restrict__ wT,
                                const float* __restrict__ b_out,
                                const float* __restrict__ feats,
                                float* __restrict__ out) {
  const int lane = threadIdx.x & 31;
  const int wave = threadIdx.x >> 5;
  const int tile = blockIdx.x * 8 + wave;     // 256*16 tiles
  const int NT = DIMF / 16;                   // 16
  const int m0 = (tile / NT) * 16;
  const int n0 = (tile % NT) * 16;
  v8f acc = {};
#pragma unroll
  for (int kk = 0; kk < INNERD; kk += 32)
    acc = wmma_bf16(load_A16x32(ao, INNERD, m0, kk, lane),
                    load_B32x16(wT, INNERD, n0, kk, lane), acc);
  const int n   = n0 + (lane & 15);
  const int mof = (lane & 16) ? 8 : 0;
  const float bb = b_out[n];
#pragma unroll
  for (int r = 0; r < 8; ++r) {
    const int m = m0 + r + mof;
    float v = gelu_erf(acc[r] + bb);
    out[(size_t)m * DIMF + n] = v + feats[(size_t)m * DIMF + n];
  }
}

// ---------------------------------------------------------------------------
// Launch
// ---------------------------------------------------------------------------
extern "C" void kernel_launch(void* const* d_in, const int* in_sizes, int n_in,
                              void* d_out, int out_size, void* d_ws, size_t ws_size,
                              hipStream_t stream) {
  (void)in_sizes; (void)n_in; (void)out_size; (void)ws_size;
  const float* xyzs    = (const float*)d_in[0];
  const float* feats   = (const float*)d_in[1];
  const float* gamma   = (const float*)d_in[2];
  const float* beta    = (const float*)d_in[3];
  const float* W_qkv   = (const float*)d_in[4];
  const float* W_pos1  = (const float*)d_in[5];
  const float* W_pos2  = (const float*)d_in[6];
  const float* W_neg1  = (const float*)d_in[7];
  const float* W_neg2  = (const float*)d_in[8];
  const float* W_cross = (const float*)d_in[9];
  const float* b_cross = (const float*)d_in[10];
  const float* W_sp    = (const float*)d_in[11];
  const float* W_out   = (const float*)d_in[12];
  const float* b_out   = (const float*)d_in[13];
  float* out = (float*)d_out;

  char* ws = (char*)d_ws;
  size_t off = 0;
  auto take = [&](size_t bytes) -> char* {
    char* p = ws + off;
    off = (off + bytes + 255) & ~(size_t)255;
    return p;
  };
  const size_t ROWS = (size_t)BATCH * NPTS;          // 4096
  bf16*  xn    = (bf16*)take(ROWS * DIMF * 2);
  bf16*  wqkvT = (bf16*)take((size_t)QKVN * DIMF * 2);
  bf16*  woutT = (bf16*)take((size_t)DIMF * INNERD * 2);
  bf16*  qb    = (bf16*)take((size_t)BATCH * HEADS * NPTS * DHEAD * 2);
  bf16*  kb    = (bf16*)take((size_t)BATCH * HEADS * NPTS * DHEAD * 2);
  bf16*  vtb   = (bf16*)take((size_t)BATCH * HEADS * DHEAD * NPTS * 2);
  bf16*  aob   = (bf16*)take(ROWS * INNERD * 2);
  float* dmod  = (float*)take((size_t)BATCH * NPTS * NPTS * 3 * 4);  // 50 MB

  cvt_wqkv_kernel<<<(QKVN * DIMF) / 256, 256, 0, stream>>>(W_qkv, wqkvT);
  cvt_wout_kernel<<<(DIMF * INNERD) / 256, 256, 0, stream>>>(W_out, woutT);
  ln_kernel<<<(int)ROWS, 256, 0, stream>>>(feats, gamma, beta, xn);
  delta_kernel<<<(BATCH * NPTS * NPTS) / 256, 256, 0, stream>>>(
      xyzs, W_pos1, W_pos2, W_neg1, W_neg2, W_cross, b_cross, dmod);
  qkv_gemm_kernel<<<(256 * 96) / 8, 256, 0, stream>>>(xn, wqkvT, qb, kb, vtb);
  attn_kernel<<<(BATCH * HEADS * 64) / 8, 256, 0, stream>>>(qb, kb, vtb, dmod, W_sp, aob);
  out_gemm_kernel<<<(256 * 16) / 8, 256, 0, stream>>>(aob, woutT, b_out, feats, out);
}
